// ASLGCN_85212151153517
// MI455X (gfx1250) — matrix operand order repaired and checked by reading, not compiled, not run on hardware
//
#include <hip/hip_runtime.h>
#include <hip/hip_bf16.h>

// ---------------------------------------------------------------------------
// GCN forward on MI455X (gfx1250, wave32).
//  - GEMMs in bf16 via v_wmma_f32_16x16x32_bf16 (VGPR layouts per CDNA5 ISA 7.12.2)
//  - A-tile staging via GLOBAL_LOAD_ASYNC_TO_LDS_B128 (+ s_wait_asynccnt)
//  - edge aggregation / BN stats / pooling via global_atomic_add_f32
// ---------------------------------------------------------------------------

#define NCH_IN   128
#define NCH_HID  256
#define NCH_OUT  64
#define N_GRAPHS 1024
#define BN_EPS   1e-5f

typedef __attribute__((ext_vector_type(16))) __bf16 v16bf;
typedef __attribute__((ext_vector_type(8)))  float  v8f;

// ------------------------------ utility kernels ----------------------------

__global__ void k_zero_f32(float* __restrict__ p, long n) {
  long i = (long)blockIdx.x * blockDim.x + threadIdx.x;
  if (i < n) p[i] = 0.0f;
}

__global__ void k_fill_f32(float* __restrict__ p, float v, long n) {
  long i = (long)blockIdx.x * blockDim.x + threadIdx.x;
  if (i < n) p[i] = v;
}

// 8 floats -> 8 bf16 per thread (n must be a multiple of 8; all our sizes are)
__global__ void k_f32_to_bf16_v8(const float* __restrict__ s, __bf16* __restrict__ d, long n8) {
  long i = (long)blockIdx.x * blockDim.x + threadIdx.x;
  if (i >= n8) return;
  const float4 a = ((const float4*)s)[i * 2 + 0];
  const float4 b = ((const float4*)s)[i * 2 + 1];
  union { __bf16 h[8]; uint4 u; } o;
  o.h[0] = (__bf16)a.x; o.h[1] = (__bf16)a.y; o.h[2] = (__bf16)a.z; o.h[3] = (__bf16)a.w;
  o.h[4] = (__bf16)b.x; o.h[5] = (__bf16)b.y; o.h[6] = (__bf16)b.z; o.h[7] = (__bf16)b.w;
  *(uint4*)(d + i * 8) = o.u;
}

// deg[col[e]] += 1  (self-loops pre-seeded with k_fill_f32(deg, 1.0))
__global__ void k_deg_edges(const int* __restrict__ col, float* __restrict__ deg, long e_cnt) {
  long i = (long)blockIdx.x * blockDim.x + threadIdx.x;
  if (i < e_cnt) atomicAdd(&deg[col[i]], 1.0f);
}

__global__ void k_rsqrt_inplace(float* __restrict__ p, long n) {
  long i = (long)blockIdx.x * blockDim.x + threadIdx.x;
  if (i < n) p[i] = rsqrtf(p[i]);
}

// --------------------------- bf16 WMMA GEMM --------------------------------
// C[M,N] = A[M,K] (bf16, row-major) * B[K,N] (bf16, row-major), f32 accum.
// Block: 256 threads = 8 waves; tile 128(M) x 64(N); K stepped by 32.
// Wave w owns rows [w*16, w*16+16); 4 accumulators cover N=64.
// A tile staged with async global->LDS DMA; out-of-range M rows are clamped
// (their WMMA results are discarded by the store guard).

#define TM 128
#define TN 64
#define TK 32
#define LDP (TK + 8)   // LDS row stride in halves (80 B -> bank rotation of 20)

__global__ __launch_bounds__(256) void k_gemm_bf16(
    const __bf16* __restrict__ A, const __bf16* __restrict__ B,
    float* __restrict__ C, int M, int N, int K)
{
  __shared__ __bf16 As[TM][LDP];
  __shared__ __bf16 Bs[TN][LDP];   // transposed: Bs[n][k]

  const int tid  = threadIdx.x;
  const int lane = tid & 31;
  const int wave = tid >> 5;               // 0..7
  const int r    = lane & 15;
  const int hi   = lane >> 4;              // 0 or 1
  const int mblk = blockIdx.x * TM;
  const int nblk = blockIdx.y * TN;

  v8f acc0 = {0,0,0,0,0,0,0,0};
  v8f acc1 = acc0, acc2 = acc0, acc3 = acc0;

  // A staging addresses (row clamped: out-of-range rows compute garbage that
  // is never stored -- row m of D depends only on row m of A).
  const int arow_st = tid >> 1;
  const int ahs     = (tid & 1) * 16;
  int aclamp = mblk + arow_st; if (aclamp > M - 1) aclamp = M - 1;
  const __bf16* aglob = A + (size_t)aclamp * K + ahs;
  // low 32 bits of a generic pointer to LDS == LDS byte offset
  const unsigned alds = (unsigned)(size_t)&As[arow_st][ahs];

  for (int k0 = 0; k0 < K; k0 += TK) {
    // ---- stage A tile via async DMA: 32 bytes/thread (IOFFSET applies to
    //      both the global and LDS side, so offset:16 reuses both bases)
    {
      const unsigned long long ga = (unsigned long long)(size_t)(aglob + k0);
      asm volatile(
          "global_load_async_to_lds_b128 %0, %1, off\n\t"
          "global_load_async_to_lds_b128 %0, %1, off offset:16"
          :: "v"(alds), "v"(ga) : "memory");
    }
    // ---- stage B tile transposed: 32(k) x 64(n); each thread moves 8 halves
    {
      const int k  = tid >> 3;
      const int ns = (tid & 7) * 8;
      uint4 v = *(const uint4*)(B + (size_t)(k0 + k) * N + nblk + ns);
      const __bf16* hv = (const __bf16*)&v;
#pragma unroll
      for (int i = 0; i < 8; ++i) Bs[ns + i][k] = hv[i];
    }
    asm volatile("s_wait_asynccnt 0x0" ::: "memory");
    __syncthreads();

    // ---- A fragment (ISA 16-bit A 16x32 layout)
    v16bf afrag;
    {
      const int arow = wave * 16 + r;
      const int ka   = hi * 8;
#pragma unroll
      for (int i = 0; i < 8; ++i) {
        afrag[i]     = As[arow][ka + i];        // K = ka .. ka+7
        afrag[8 + i] = As[arow][16 + ka + i];   // K = 16+ka .. 16+ka+7
      }
    }
    // ---- 4 B fragments + WMMA (B: col = lane&15, K = hi*8 .. hi*8+7, 16+hi*8 .. 16+hi*8+7)
    const int kb = hi * 8;
#pragma unroll
    for (int nt = 0; nt < 4; ++nt) {
      v16bf bfrag;
      const int bn = nt * 16 + r;
#pragma unroll
      for (int i = 0; i < 8; ++i) { bfrag[i] = Bs[bn][kb + i]; bfrag[8 + i] = Bs[bn][16 + kb + i]; }
      v8f* accp = (nt == 0) ? &acc0 : (nt == 1) ? &acc1 : (nt == 2) ? &acc2 : &acc3;
      *accp = __builtin_amdgcn_wmma_f32_16x16x32_bf16(
          false, afrag, false, bfrag, (short)0, *accp, false, false);
    }
    __syncthreads();
  }

  // ---- store D (ISA 32-bit C/D 16x16 layout: VGPR v -> M = v + 8*hi, N = lane&15)
  // Fast path is block-uniform (only the last M-block is partial), so the
  // common case emits plain affine stores with no per-element exec masking.
  const int gm0 = mblk + wave * 16 + 8 * hi;           // first of 8 rows
  float* crow = C + (size_t)gm0 * N + nblk + r;        // per-lane row pointer
  if (mblk + TM <= M) {
#pragma unroll
    for (int v = 0; v < 8; ++v) {
      float* p = crow + (size_t)v * N;
      p[0]  = acc0[v];
      p[16] = acc1[v];
      p[32] = acc2[v];
      p[48] = acc3[v];
    }
  } else {
#pragma unroll
    for (int v = 0; v < 8; ++v) {
      if (gm0 + v < M) {
        float* p = crow + (size_t)v * N;
        p[0]  = acc0[v];
        p[16] = acc1[v];
        p[32] = acc2[v];
        p[48] = acc3[v];
      }
    }
  }
}

// ----------------------- GCN aggregation kernels ---------------------------

// agg[i][c] = dinv[i]^2 * xw[i][c] + bias[c]   (self-loop contribution + bias)
// float4 over C (C multiple of 4)
__global__ void k_agg_init_v4(const float4* __restrict__ xw4, const float* __restrict__ dinv,
                              const float4* __restrict__ bias4, float4* __restrict__ agg4,
                              int nn, int C4) {
  long i = (long)blockIdx.x * blockDim.x + threadIdx.x;
  if (i >= (long)nn * C4) return;
  int rr = (int)(i / C4), c4 = (int)(i % C4);
  float d = dinv[rr];
  float dd = d * d;
  float4 v = xw4[i];
  float4 b = bias4[c4];
  float4 o;
  o.x = dd * v.x + b.x; o.y = dd * v.y + b.y;
  o.z = dd * v.z + b.z; o.w = dd * v.w + b.w;
  agg4[i] = o;
}

// 4 edges per 256-thread block; 64 lanes/edge, float4 per lane:
// agg[col][c] += dinv[row]*dinv[col] * xw[row][c]
__global__ __launch_bounds__(256) void k_agg_edges(
    const int* __restrict__ row, const int* __restrict__ col,
    const float* __restrict__ dinv, const float* __restrict__ xw,
    float* __restrict__ agg, int C, long E) {
  const long e = (long)blockIdx.x * 4 + (threadIdx.x >> 6);
  if (e >= E) return;
  const int q = threadIdx.x & 63;          // 64 lanes x float4 == 256 channels
  const int rr = row[e];
  const int tt = col[e];
  const float nrm = dinv[rr] * dinv[tt];
  const float4 v = *(const float4*)(xw + (size_t)rr * C + q * 4);
  float* dst = agg + (size_t)tt * C + q * 4;
  atomicAdd(dst + 0, nrm * v.x);
  atomicAdd(dst + 1, nrm * v.y);
  atomicAdd(dst + 2, nrm * v.z);
  atomicAdd(dst + 3, nrm * v.w);
}

// ------------------------------ batch norm ---------------------------------

__global__ __launch_bounds__(NCH_HID) void k_bn_stats(
    const float* __restrict__ agg, float* __restrict__ sum, float* __restrict__ sumsq,
    int nn, int C) {
  const int c  = threadIdx.x;
  const int r0 = blockIdx.x * 128;
  int rend = r0 + 128; if (rend > nn) rend = nn;
  float s = 0.f, q = 0.f;
  for (int rr = r0; rr < rend; ++rr) {
    float v = agg[(size_t)rr * C + c];
    s += v; q += v * v;
  }
  atomicAdd(&sum[c], s);
  atomicAdd(&sumsq[c], q);
}

__global__ __launch_bounds__(NCH_HID) void k_bn_final(
    const float* __restrict__ sum, const float* __restrict__ sumsq,
    const float* __restrict__ gamma, const float* __restrict__ beta,
    float* __restrict__ scale, float* __restrict__ shift, int nn) {
  const int c = threadIdx.x;
  const float invN = 1.0f / (float)nn;
  const float mean = sum[c] * invN;
  const float var  = sumsq[c] * invN - mean * mean;
  const float sc   = gamma[c] * rsqrtf(var + BN_EPS);
  scale[c] = sc;
  shift[c] = beta[c] - mean * sc;
}

// h = relu(agg*scale + shift); write f32 in place + bf16 copy for next GEMM
__global__ void k_bn_apply_relu_v4(float4* __restrict__ agg4, __bf16* __restrict__ hbf,
                                   const float4* __restrict__ scale4,
                                   const float4* __restrict__ shift4, int nn, int C4) {
  long i = (long)blockIdx.x * blockDim.x + threadIdx.x;
  if (i >= (long)nn * C4) return;
  int c4 = (int)(i % C4);
  float4 v = agg4[i];
  float4 sc = scale4[c4], sh = shift4[c4];
  float4 o;
  o.x = fmaxf(v.x * sc.x + sh.x, 0.f);
  o.y = fmaxf(v.y * sc.y + sh.y, 0.f);
  o.z = fmaxf(v.z * sc.z + sh.z, 0.f);
  o.w = fmaxf(v.w * sc.w + sh.w, 0.f);
  agg4[i] = o;
  union { __bf16 h[4]; uint2 u; } pk;
  pk.h[0] = (__bf16)o.x; pk.h[1] = (__bf16)o.y; pk.h[2] = (__bf16)o.z; pk.h[3] = (__bf16)o.w;
  *(uint2*)(hbf + i * 4) = pk.u;
}

// ------------------------------ pooling ------------------------------------

// 4 rows per 256-thread block; 64 lanes/row, float4 per lane
__global__ __launch_bounds__(256) void k_pool(
    const int* __restrict__ batch, const float* __restrict__ h,
    float* __restrict__ pooled, float* __restrict__ counts, int C, int nn) {
  const long rr = (long)blockIdx.x * 4 + (threadIdx.x >> 6);
  if (rr >= nn) return;
  const int q = threadIdx.x & 63;
  const int g = batch[rr];
  const float4 v = *(const float4*)(h + (size_t)rr * C + q * 4);
  float* dst = pooled + (size_t)g * C + q * 4;
  atomicAdd(dst + 0, v.x);
  atomicAdd(dst + 1, v.y);
  atomicAdd(dst + 2, v.z);
  atomicAdd(dst + 3, v.w);
  if (q == 0) atomicAdd(&counts[g], 1.0f);
}

__global__ void k_pool_div_bf16(const float* __restrict__ pooled,
                                const float* __restrict__ counts,
                                __bf16* __restrict__ pbf, int C, long n) {
  long i = (long)blockIdx.x * blockDim.x + threadIdx.x;
  if (i >= n) return;
  int g = (int)(i / C);
  pbf[i] = (__bf16)(pooled[i] / fmaxf(counts[g], 1.0f));
}

__global__ void k_bias_add(float* __restrict__ out, const float* __restrict__ bias,
                           int C, long n) {
  long i = (long)blockIdx.x * blockDim.x + threadIdx.x;
  if (i < n) out[i] += bias[(int)(i % C)];
}

// ------------------------------ launcher -----------------------------------

extern "C" void kernel_launch(void* const* d_in, const int* in_sizes, int n_in,
                              void* d_out, int out_size, void* d_ws, size_t ws_size,
                              hipStream_t stream) {
  const float* x    = (const float*)d_in[0];
  const int*   ei   = (const int*)d_in[1];    // [2, E]
  const int*   bat  = (const int*)d_in[2];    // [NN]
  const float* W1   = (const float*)d_in[3];
  const float* b1   = (const float*)d_in[4];
  const float* g1   = (const float*)d_in[5];
  const float* be1  = (const float*)d_in[6];
  const float* W2   = (const float*)d_in[7];
  const float* b2   = (const float*)d_in[8];
  const float* g2   = (const float*)d_in[9];
  const float* be2  = (const float*)d_in[10];
  const float* Wl   = (const float*)d_in[11];
  const float* bl   = (const float*)d_in[12];
  float* out = (float*)d_out;

  const int NN = in_sizes[0] / NCH_IN;
  const int E  = in_sizes[1] / 2;
  const int* erow = ei;
  const int* ecol = ei + E;

  // ---- carve workspace
  uintptr_t base = (uintptr_t)d_ws;
  size_t off = 0;
  auto carve = [&](size_t bytes) -> void* {
    off = (off + 255) & ~(size_t)255;
    void* p = (void*)(base + off);
    off += bytes;
    return p;
  };
  float*  deg    = (float*) carve((size_t)NN * 4);                 // -> dinv (in place)
  __bf16* x_bf   = (__bf16*)carve((size_t)NN * NCH_IN * 2);
  __bf16* h_bf   = (__bf16*)carve((size_t)NN * NCH_HID * 2);
  __bf16* w1_bf  = (__bf16*)carve((size_t)NCH_IN * NCH_HID * 2);
  __bf16* w2_bf  = (__bf16*)carve((size_t)NCH_HID * NCH_HID * 2);
  __bf16* wl_bf  = (__bf16*)carve((size_t)NCH_HID * NCH_OUT * 2);
  float*  xw     = (float*) carve((size_t)NN * NCH_HID * 4);
  float*  agg    = (float*) carve((size_t)NN * NCH_HID * 4);
  float*  bnsum  = (float*) carve(NCH_HID * 4);
  float*  bnsq   = (float*) carve(NCH_HID * 4);
  float*  scale  = (float*) carve(NCH_HID * 4);
  float*  shift  = (float*) carve(NCH_HID * 4);
  float*  pooled = (float*) carve((size_t)N_GRAPHS * NCH_HID * 4);
  float*  counts = (float*) carve((size_t)N_GRAPHS * 4);
  __bf16* p_bf   = (__bf16*)carve((size_t)N_GRAPHS * NCH_HID * 2);

  const int B = 256;
  auto blocks = [](long n, int b) { return (unsigned)((n + b - 1) / b); };
  const int C4 = NCH_HID / 4;

  // ---- precision conversion (bf16 operands for WMMA)
  k_f32_to_bf16_v8<<<blocks((long)NN * NCH_IN / 8, B), B, 0, stream>>>(x, x_bf, (long)NN * NCH_IN / 8);
  k_f32_to_bf16_v8<<<blocks(NCH_IN * NCH_HID / 8, B), B, 0, stream>>>(W1, w1_bf, NCH_IN * NCH_HID / 8);
  k_f32_to_bf16_v8<<<blocks(NCH_HID * NCH_HID / 8, B), B, 0, stream>>>(W2, w2_bf, NCH_HID * NCH_HID / 8);
  k_f32_to_bf16_v8<<<blocks(NCH_HID * NCH_OUT / 8, B), B, 0, stream>>>(Wl, wl_bf, NCH_HID * NCH_OUT / 8);

  // ---- degrees (self-loops included), then deg <- rsqrt(deg)
  k_fill_f32<<<blocks(NN, B), B, 0, stream>>>(deg, 1.0f, NN);
  k_deg_edges<<<blocks(E, B), B, 0, stream>>>(ecol, deg, E);
  k_rsqrt_inplace<<<blocks(NN, B), B, 0, stream>>>(deg, NN);
  float* dinv = deg;

  dim3 gemmBlk(256);
  dim3 g1grid((NN + TM - 1) / TM, NCH_HID / TN);
  dim3 g3grid(N_GRAPHS / TM, NCH_OUT / TN);

  // =================== layer 1 ===================
  k_gemm_bf16<<<g1grid, gemmBlk, 0, stream>>>(x_bf, w1_bf, xw, NN, NCH_HID, NCH_IN);
  k_agg_init_v4<<<blocks((long)NN * C4, B), B, 0, stream>>>(
      (const float4*)xw, dinv, (const float4*)b1, (float4*)agg, NN, C4);
  k_agg_edges<<<blocks(E, 4), 256, 0, stream>>>(erow, ecol, dinv, xw, agg, NCH_HID, E);
  k_zero_f32<<<blocks(NCH_HID, B), B, 0, stream>>>(bnsum, NCH_HID);
  k_zero_f32<<<blocks(NCH_HID, B), B, 0, stream>>>(bnsq, NCH_HID);
  k_bn_stats<<<blocks(NN, 128), NCH_HID, 0, stream>>>(agg, bnsum, bnsq, NN, NCH_HID);
  k_bn_final<<<1, NCH_HID, 0, stream>>>(bnsum, bnsq, g1, be1, scale, shift, NN);
  k_bn_apply_relu_v4<<<blocks((long)NN * C4, B), B, 0, stream>>>(
      (float4*)agg, h_bf, (const float4*)scale, (const float4*)shift, NN, C4);

  // =================== layer 2 ===================
  k_gemm_bf16<<<g1grid, gemmBlk, 0, stream>>>(h_bf, w2_bf, xw, NN, NCH_HID, NCH_HID);
  k_agg_init_v4<<<blocks((long)NN * C4, B), B, 0, stream>>>(
      (const float4*)xw, dinv, (const float4*)b2, (float4*)agg, NN, C4);
  k_agg_edges<<<blocks(E, 4), 256, 0, stream>>>(erow, ecol, dinv, xw, agg, NCH_HID, E);
  k_zero_f32<<<blocks(NCH_HID, B), B, 0, stream>>>(bnsum, NCH_HID);
  k_zero_f32<<<blocks(NCH_HID, B), B, 0, stream>>>(bnsq, NCH_HID);
  k_bn_stats<<<blocks(NN, 128), NCH_HID, 0, stream>>>(agg, bnsum, bnsq, NN, NCH_HID);
  k_bn_final<<<1, NCH_HID, 0, stream>>>(bnsum, bnsq, g2, be2, scale, shift, NN);
  k_bn_apply_relu_v4<<<blocks((long)NN * C4, B), B, 0, stream>>>(
      (float4*)agg, h_bf, (const float4*)scale, (const float4*)shift, NN, C4);

  // =================== pooling + head ===================
  k_zero_f32<<<blocks((long)N_GRAPHS * NCH_HID, B), B, 0, stream>>>(pooled, (long)N_GRAPHS * NCH_HID);
  k_zero_f32<<<blocks(N_GRAPHS, B), B, 0, stream>>>(counts, N_GRAPHS);
  k_pool<<<blocks(NN, 4), 256, 0, stream>>>(bat, agg, pooled, counts, NCH_HID, NN);
  k_pool_div_bf16<<<blocks((long)N_GRAPHS * NCH_HID, B), B, 0, stream>>>(
      pooled, counts, p_bf, NCH_HID, (long)N_GRAPHS * NCH_HID);
  k_gemm_bf16<<<g3grid, gemmBlk, 0, stream>>>(p_bf, wl_bf, out, N_GRAPHS, NCH_OUT, NCH_HID);
  k_bias_add<<<blocks((long)N_GRAPHS * NCH_OUT, B), B, 0, stream>>>(
      out, bl, NCH_OUT, (long)N_GRAPHS * NCH_OUT);
}
